// EPCQ_39281770889637
// MI455X (gfx1250) — matrix operand order repaired and checked by reading, not compiled, not run on hardware
//
#include <hip/hip_runtime.h>
#include <hip/hip_bf16.h>
#include <math.h>

// ---- problem constants (match reference) ----
#define N_PTS     500000
#define FEAT      32
#define R_RAYS    4096
#define S_SAMP    24
#define K_NB      8
#define RS        (R_RAYS * S_SAMP)          // 98304 samples
#define EPSV      1e-8f
#define FAR_DELTA 100.0f
#define HPAD      64                          // K padded: 32 feat + 3 rd + zeros

typedef __attribute__((ext_vector_type(16))) _Float16 v16h;
typedef __attribute__((ext_vector_type(8)))  _Float16 v8h;
typedef __attribute__((ext_vector_type(8)))  float    v8f;

__device__ __forceinline__ float sigmoidf_(float x) {
    return 1.0f / (1.0f + __expf(-x));
}
__device__ __forceinline__ float softplusf_(float x) {
    return (x > 20.0f) ? x : log1pf(__expf(x));
}

// =====================================================================
// Phase 1: per-sample neighbor gather + inverse-distance aggregation.
// One thread per (ray, sample). Writes:
//   Hws      [RS][64] f16  : [agg_feat(32), rd(3), zeros] (WMMA A operand)
//   sigma_ws [RS]     f32  : softplus(agg_dens) * mask
//   rgb_ws   [RS][3]  f32  : agg_rgb (added to head logits later)
// =====================================================================
__global__ void pn_aggregate(const float* __restrict__ pc_xyz,
                             const float* __restrict__ pc_rgb,
                             const float* __restrict__ pc_dir,
                             const float* __restrict__ pc_feat,
                             const float* __restrict__ pc_density,
                             const float* __restrict__ sp_xyz,
                             const float* __restrict__ rays_d,
                             const int*   __restrict__ pc_idx,
                             const int*   __restrict__ mask,
                             _Float16*    __restrict__ Hws,
                             float*       __restrict__ sigma_ws,
                             float*       __restrict__ rgb_ws)
{
    int rs = blockIdx.x * blockDim.x + threadIdx.x;
    if (rs >= RS) return;
    int r = rs / S_SAMP;

    // normalized ray direction
    float dx = rays_d[r * 3 + 0], dy = rays_d[r * 3 + 1], dz = rays_d[r * 3 + 2];
    float rinv = 1.0f / (sqrtf(dx * dx + dy * dy + dz * dz) + EPSV);
    float rdx = dx * rinv, rdy = dy * rinv, rdz = dz * rinv;

    float sx = sp_xyz[rs * 3 + 0];
    float sy = sp_xyz[rs * 3 + 1];
    float sz = sp_xyz[rs * 3 + 2];

    // preload indices, prefetch 128B feature rows (gfx1250 global_prefetch_b8)
    int nb[K_NB];
    const int* idxp = pc_idx + (size_t)rs * K_NB;
#pragma unroll
    for (int k = 0; k < K_NB; ++k) nb[k] = idxp[k];
#pragma unroll
    for (int k = 0; k < K_NB; ++k)
        __builtin_prefetch(pc_feat + (size_t)nb[k] * FEAT, 0, 3);

    float accf[FEAT];
#pragma unroll
    for (int f = 0; f < FEAT; ++f) accf[f] = 0.0f;
    float ar = 0.f, ag = 0.f, ab = 0.f, adens = 0.f, wsum = 0.f;

#pragma unroll
    for (int k = 0; k < K_NB; ++k) {
        int i = nb[k];
        const float* px = pc_xyz + (size_t)i * 3;
        const float* pd = pc_dir + (size_t)i * 3;
        const float* pr = pc_rgb + (size_t)i * 3;

        float ddx = px[0] - sx, ddy = px[1] - sy, ddz = px[2] - sz;
        float dist = sqrtf(ddx * ddx + ddy * ddy + ddz * ddz + EPSV);
        float dot  = pd[0] * rdx + pd[1] * rdy + pd[2] * rdz;
        float w    = (1.0f / dist) * sigmoidf_(dot);
        wsum += w;

        const float4* frow = (const float4*)(pc_feat + (size_t)i * FEAT);
#pragma unroll
        for (int c = 0; c < FEAT / 4; ++c) {
            float4 f4 = frow[c];
            accf[4 * c + 0] += w * f4.x;
            accf[4 * c + 1] += w * f4.y;
            accf[4 * c + 2] += w * f4.z;
            accf[4 * c + 3] += w * f4.w;
        }
        ar += w * pr[0];
        ag += w * pr[1];
        ab += w * pr[2];
        adens += w * pc_density[i];
    }

    float invw = 1.0f / (wsum + EPSV);

    // H row (f16, padded to 64): feats, rd, zeros — stored as 8x b128
    _Float16 hbuf[HPAD];
#pragma unroll
    for (int f = 0; f < FEAT; ++f) hbuf[f] = (_Float16)(accf[f] * invw);
    hbuf[32] = (_Float16)rdx;
    hbuf[33] = (_Float16)rdy;
    hbuf[34] = (_Float16)rdz;
#pragma unroll
    for (int f = 35; f < HPAD; ++f) hbuf[f] = (_Float16)0.0f;

    _Float16* hrow = Hws + (size_t)rs * HPAD;
#pragma unroll
    for (int c = 0; c < HPAD / 8; ++c)
        *(v8h*)(hrow + 8 * c) = *(const v8h*)(hbuf + 8 * c);

    rgb_ws[rs * 3 + 0] = ar * invw;
    rgb_ws[rs * 3 + 1] = ag * invw;
    rgb_ws[rs * 3 + 2] = ab * invw;

    float m = (float)mask[rs];
    sigma_ws[rs] = softplusf_(adens * invw) * m;
}

// =====================================================================
// Phase 2: color head as WMMA GEMM.
// D[RS,16] = H[RS,64] x Wpad[64,16]  (only N<3 columns are real).
// One wave per 16 rows; K=64 via two v_wmma_f32_16x16x32_f16 steps.
//
// The padded weight matrix is a shared 2KB constant: stage it ONCE per
// block in LDS, *transposed* (Wt[n][k]) so each lane's 16 B-halves per
// K-step are contiguous -> two aligned ds_load_b128, no divergence.
//
// A layout (16-bit A 16x32, as loaded here): lanes 0-15 = rows M; halves
//   0..15 -> K = kbase + khi*16 + e (contiguous, same order as B)
//   (khi = lane/16 adds +16). Two global b128 loads per step.
// B layout (32x16): lane%16 = column N, lane/16 = K slab of 16;
//   half e -> K = kbase + khi*16 + e  ->  Wt[nrow][kbase + khi*16 + e].
// =====================================================================
__global__ void pn_head_wmma(const float* __restrict__ W_color,   // [35,3]
                             const float* __restrict__ b_color,   // [3]
                             const _Float16* __restrict__ Hws,    // [RS,64]
                             const float* __restrict__ rgb_ws,    // [RS,3]
                             float* __restrict__ color_ws)        // [RS,3]
{
    __shared__ _Float16 Wt[16][HPAD];   // transposed padded weights, 2KB

    // cooperative fill: 1024 halves, pad predicate applied once per elem
    for (int idx = threadIdx.x; idx < 16 * HPAD; idx += blockDim.x) {
        int n = idx >> 6;        // 0..15
        int k = idx & (HPAD - 1);
        Wt[n][k] = (_Float16)((k < FEAT + 3 && n < 3) ? W_color[k * 3 + n] : 0.0f);
    }
    __syncthreads();

    int tid  = blockIdx.x * blockDim.x + threadIdx.x;
    int wave = tid >> 5;
    int lane = tid & 31;
    int m0   = wave * 16;            // first sample row of this tile
    int nrow = lane & 15;
    int khi  = (lane >> 4) & 1;

    const _Float16* hrow = Hws + (size_t)(m0 + nrow) * HPAD;
    const _Float16* wrow = &Wt[nrow][khi * 16];

    // B operands hoisted out of the K loop: 16 contiguous halves each
    v16h b0 = *(const v16h*)(wrow);        // kstep 0: K = khi*16 .. +15
    v16h b1 = *(const v16h*)(wrow + 32);   // kstep 1: K = 32+khi*16 .. +15

    v8f c = {};
#pragma unroll
    for (int ks = 0; ks < 2; ++ks) {
        const int kbase = ks * 32;

        // ---- A operand: 16 contiguous halves, K = kbase + khi*16 .. +15 ----
        v16h a = *(const v16h*)(hrow + kbase + khi * 16);

        c = __builtin_amdgcn_wmma_f32_16x16x32_f16(
                /*neg_a=*/false, a, /*neg_b=*/false, (ks == 0) ? b0 : b1,
                /*c_mod=*/(short)0, c, /*reuse_a=*/false, /*reuse_b=*/false);
    }

    // D layout: VGPR v -> M = v + khi*8, N = lane%16
    if (nrow < 3) {
        float bc = b_color[nrow];
#pragma unroll
        for (int v = 0; v < 8; ++v) {
            int row = m0 + v + khi * 8;
            float logit = c[v] + bc + rgb_ws[row * 3 + nrow];
            color_ws[row * 3 + nrow] = sigmoidf_(logit);
        }
    }
}

// =====================================================================
// Phase 3: per-ray volumetric composite (sequential S=24 scan).
// =====================================================================
__global__ void pn_composite(const float* __restrict__ sp_t,      // [R,S]
                             const float* __restrict__ sigma_ws,  // [RS]
                             const float* __restrict__ color_ws,  // [RS,3]
                             float* __restrict__ out)             // [R,3]
{
    int r = blockIdx.x * blockDim.x + threadIdx.x;
    if (r >= R_RAYS) return;

    float T = 1.0f, acc = 0.0f, cr = 0.0f, cg = 0.0f, cb = 0.0f;
    float tc = fmaxf(sp_t[r * S_SAMP + 0], 0.0f);
#pragma unroll
    for (int s = 0; s < S_SAMP; ++s) {
        float tn    = (s + 1 < S_SAMP) ? fmaxf(sp_t[r * S_SAMP + s + 1], 0.0f) : 0.0f;
        float delta = (s + 1 < S_SAMP) ? (tn - tc) : FAR_DELTA;
        int rs = r * S_SAMP + s;

        float sg    = sigma_ws[rs];
        float alpha = 1.0f - __expf(-sg * delta);
        float w     = T * alpha;
        T *= (1.0f - alpha + 1e-10f);

        cr  += w * color_ws[rs * 3 + 0];
        cg  += w * color_ws[rs * 3 + 1];
        cb  += w * color_ws[rs * 3 + 2];
        acc += w;
        tc = tn;
    }
    float bg = 1.0f - acc;
    out[r * 3 + 0] = cr + bg;
    out[r * 3 + 1] = cg + bg;
    out[r * 3 + 2] = cb + bg;
}

// =====================================================================
extern "C" void kernel_launch(void* const* d_in, const int* in_sizes, int n_in,
                              void* d_out, int out_size, void* d_ws, size_t ws_size,
                              hipStream_t stream) {
    (void)in_sizes; (void)n_in; (void)out_size; (void)ws_size;

    const float* pc_xyz     = (const float*)d_in[0];
    const float* pc_rgb     = (const float*)d_in[1];
    const float* pc_dir     = (const float*)d_in[2];
    const float* pc_feat    = (const float*)d_in[3];
    const float* pc_density = (const float*)d_in[4];
    const float* W_color    = (const float*)d_in[5];
    const float* b_color    = (const float*)d_in[6];
    const float* sp_xyz     = (const float*)d_in[7];
    const float* sp_t       = (const float*)d_in[8];
    const float* rays_d     = (const float*)d_in[9];
    const int*   pc_idx     = (const int*)d_in[10];
    const int*   mask       = (const int*)d_in[11];
    float*       out        = (float*)d_out;

    // workspace layout (all 16B aligned)
    char* ws = (char*)d_ws;
    size_t offH     = 0;
    size_t offSigma = offH + (size_t)RS * HPAD * sizeof(_Float16);  // 12.58 MB
    size_t offRgb   = offSigma + (size_t)RS * sizeof(float);
    size_t offColor = offRgb + (size_t)RS * 3 * sizeof(float);
    _Float16* Hws      = (_Float16*)(ws + offH);
    float*    sigma_ws = (float*)(ws + offSigma);
    float*    rgb_ws   = (float*)(ws + offRgb);
    float*    color_ws = (float*)(ws + offColor);

    // Phase 1: 98304 samples, 256 threads/block
    pn_aggregate<<<(RS + 255) / 256, 256, 0, stream>>>(
        pc_xyz, pc_rgb, pc_dir, pc_feat, pc_density,
        sp_xyz, rays_d, pc_idx, mask, Hws, sigma_ws, rgb_ws);

    // Phase 2: 6144 waves (16 rows each) -> 196608 threads
    pn_head_wmma<<<(RS / 16) * 32 / 256, 256, 0, stream>>>(
        W_color, b_color, Hws, rgb_ws, color_ws);

    // Phase 3: one thread per ray
    pn_composite<<<(R_RAYS + 255) / 256, 256, 0, stream>>>(
        sp_t, sigma_ws, color_ws, out);
}
